// Encoder_4715874091274
// MI455X (gfx1250) — matrix operand order repaired and checked by reading, not compiled, not run on hardware
//
#include <hip/hip_runtime.h>
#include <hip/hip_bf16.h>
#include <stdint.h>

// ---------------- types ----------------
typedef __bf16 v16bf __attribute__((ext_vector_type(16)));
typedef float  v8f   __attribute__((ext_vector_type(8)));
typedef unsigned int u32x4 __attribute__((ext_vector_type(4)));

// fp32 -> bf16 (round to nearest even)
static __device__ __forceinline__ unsigned short f2bf(float f) {
    union { float f; uint32_t u; } c; c.f = f;
    uint32_t u = c.u;
    u = (u + 0x7fffu + ((u >> 16) & 1u)) >> 16;
    return (unsigned short)u;
}
static __device__ __forceinline__ uint32_t pack2(float f0, float f1) {
    return (uint32_t)f2bf(f0) | ((uint32_t)f2bf(f1) << 16);
}

union Frag { v16bf v; uint32_t u[8]; u32x4 q[2]; };

// ---------------- generic WMMA GEMM, 32x64 strip per wave ----------------
// C[M,N] = A[M,K] * B[K,N] (+bias)(+resid)(relu), per-z batch offsets.
// AMODE: 0 = bf16 row-major [M,K], 1 = fp32 row-major [M,K] (converted in-reg)
// B is ALWAYS bf16 [N,K] (B^T row-major, contiguous along K) -> two b128 loads/frag.
// Each wave: 2 A-fragments x 4 B-fragments -> 8 v_wmma per K-step,
// 12 b128 loads per 8 WMMAs (B reused across both M-tiles).
template<int AMODE, bool BIAS, bool RELU, bool RESID>
__global__ void gemm_ws(const void* __restrict__ A, int lda, long long sA1, long long sA2,
                        const unsigned short* __restrict__ Bt, int ldb, long long sB1, long long sB2,
                        float* __restrict__ C, int ldc, long long sC1, long long sC2,
                        unsigned short* __restrict__ Cb, int ldcb, long long sCb1, long long sCb2,
                        const float* __restrict__ bias,
                        const float* __restrict__ resid, int ldr,
                        int M, int N, int K, int Z2) {
    const int lane = threadIdx.x & 31;
    const int wave = threadIdx.x >> 5;
    const int lm   = lane & 15;
    const int half = lane >> 4;

    const int ng = N >> 6;  // 64-wide column groups
    int tile = blockIdx.x * (blockDim.x >> 5) + wave;
    const int tm = (tile / ng) << 5;   // 32-row strip
    const int tn = (tile % ng) << 6;   // 64-col strip

    const int z  = blockIdx.z;
    const int z1 = z / Z2;
    const int z2 = z - z1 * Z2;
    const long long offA  = (long long)z1 * sA1  + (long long)z2 * sA2;
    const long long offB  = (long long)z1 * sB1  + (long long)z2 * sB2;
    const long long offC  = (long long)z1 * sC1  + (long long)z2 * sC2;
    const long long offCb = (long long)z1 * sCb1 + (long long)z2 * sCb2;

    v8f acc[2][4] = {{{}, {}, {}, {}}, {{}, {}, {}, {}}};
    const unsigned short* Brow = Bt + offB + (long long)(tn + lm) * ldb;

#pragma unroll 2
    for (int kk = 0; kk < K; kk += 32) {
        Frag ua[2];
        if (AMODE == 0) {
            const unsigned short* Ab =
                (const unsigned short*)A + offA + (long long)(tm + lm) * lda;
#pragma unroll
            for (int mi = 0; mi < 2; ++mi) {
                const unsigned short* Ap = Ab + (long long)(mi * 16) * lda;
                ua[mi].q[0] = *(const u32x4*)(Ap + kk + half * 8);
                ua[mi].q[1] = *(const u32x4*)(Ap + kk + 16 + half * 8);
            }
        } else {
            const float* Af = (const float*)A + offA + (long long)(tm + lm) * lda;
#pragma unroll
            for (int mi = 0; mi < 2; ++mi) {
                const float* Ap = Af + (long long)(mi * 16) * lda;
#pragma unroll
                for (int j = 0; j < 8; ++j) {
                    int kb = kk + ((j < 4) ? 0 : 16) + half * 8 + ((j & 3) << 1);
                    ua[mi].u[j] = pack2(Ap[kb], Ap[kb + 1]);
                }
            }
        }
#pragma unroll
        for (int t = 0; t < 4; ++t) {
            Frag ub;
            const unsigned short* Bp = Brow + (long long)(t * 16) * ldb + kk + half * 8;
            ub.q[0] = *(const u32x4*)(Bp);
            ub.q[1] = *(const u32x4*)(Bp + 16);
            acc[0][t] = __builtin_amdgcn_wmma_f32_16x16x32_bf16(false, ua[0].v, false, ub.v,
                                                                (short)0, acc[0][t], false, false);
            acc[1][t] = __builtin_amdgcn_wmma_f32_16x16x32_bf16(false, ua[1].v, false, ub.v,
                                                                (short)0, acc[1][t], false, false);
        }
    }

    // ---- epilogue: D element (m,n): lane = n%16 + 16*(m>=8), vgpr = m%8
    float bv[4];
    if (BIAS) {
#pragma unroll
        for (int t = 0; t < 4; ++t) bv[t] = bias[tn + t * 16 + lm];
    }
#pragma unroll
    for (int mi = 0; mi < 2; ++mi) {
#pragma unroll
        for (int r = 0; r < 8; ++r) {
            const long long row = tm + mi * 16 + half * 8 + r;
#pragma unroll
            for (int t = 0; t < 4; ++t) {
                const int coln = tn + t * 16 + lm;
                float v = acc[mi][t][r];
                if (BIAS)  v += bv[t];
                if (RESID) v += resid[row * ldr + coln];
                if (RELU)  v = fmaxf(v, 0.f);
                if (C)  C[offC + row * ldc + coln] = v;
                if (Cb) Cb[offCb + row * ldcb + coln] = f2bf(v);
            }
        }
    }
}

// ---------------- weight transpose+convert: fp32 [K,N] -> bf16 [N,K] ----------------
__global__ void wtr_kernel(const float* __restrict__ in, unsigned short* __restrict__ out,
                           int K, int N, long long sIn, long long sOut) {
    long long i = (long long)blockIdx.x * blockDim.x + threadIdx.x;  // < K*N
    const float* ip = in + (long long)blockIdx.z * sIn;
    unsigned short* op = out + (long long)blockIdx.z * sOut;
    int k = (int)(i / N), n = (int)(i - (long long)k * N);
    op[(long long)n * K + k] = f2bf(ip[i]);
}

// ---------------- V transpose: bf16 [B*T, 256] -> Vt [B,H,64,T] ----------------
__global__ void vt_kernel(const unsigned short* __restrict__ V, unsigned short* __restrict__ Vt) {
    long long i = (long long)blockIdx.x * blockDim.x + threadIdx.x;  // < B*T*256
    int c = (int)(i & 255);       // h*64 + e
    long long bt = i >> 8;        // b*1024 + t
    int b = (int)(bt >> 10);
    int t = (int)(bt & 1023);
    int h = c >> 6, e = c & 63;
    Vt[(((long long)(b * 4 + h) * 64 + e) << 10) + t] = V[i];
}

// ---------------- LayerNorm over D=256, one row per block ----------------
template<bool OUTBF>
__global__ void ln_kernel(const float* __restrict__ x, const float* __restrict__ g,
                          const float* __restrict__ b, void* __restrict__ out) {
    __shared__ float sred[256];
    const long long row = blockIdx.x;
    const int tid = threadIdx.x;
    float v = x[row * 256 + tid];
    sred[tid] = v; __syncthreads();
    for (int s = 128; s > 0; s >>= 1) {
        if (tid < s) sred[tid] += sred[tid + s];
        __syncthreads();
    }
    float mu = sred[0] * (1.f / 256.f);
    __syncthreads();
    float d = v - mu;
    sred[tid] = d * d; __syncthreads();
    for (int s = 128; s > 0; s >>= 1) {
        if (tid < s) sred[tid] += sred[tid + s];
        __syncthreads();
    }
    float var = sred[0] * (1.f / 256.f);
    float y = d * rsqrtf(var + 1e-5f) * g[tid] + b[tid];
    if (OUTBF) ((unsigned short*)out)[row * 256 + tid] = f2bf(y);
    else       ((float*)out)[row * 256 + tid] = y;
}

// ---------------- in-place row softmax (row length 1024), scale folded ----------------
__global__ void softmax_kernel(float* __restrict__ p, float scale) {
    __shared__ float sred[256];
    float* r = p + (long long)blockIdx.x * 1024;
    const int tid = threadIdx.x;
    float v0 = r[tid], v1 = r[tid + 256], v2 = r[tid + 512], v3 = r[tid + 768];
    float m = fmaxf(fmaxf(v0, v1), fmaxf(v2, v3));
    sred[tid] = m; __syncthreads();
    for (int s = 128; s > 0; s >>= 1) {
        if (tid < s) sred[tid] = fmaxf(sred[tid], sred[tid + s]);
        __syncthreads();
    }
    m = sred[0]; __syncthreads();
    float e0 = __expf((v0 - m) * scale);
    float e1 = __expf((v1 - m) * scale);
    float e2 = __expf((v2 - m) * scale);
    float e3 = __expf((v3 - m) * scale);
    sred[tid] = e0 + e1 + e2 + e3; __syncthreads();
    for (int s = 128; s > 0; s >>= 1) {
        if (tid < s) sred[tid] += sred[tid + s];
        __syncthreads();
    }
    float inv = 1.f / sred[0];
    r[tid] = e0 * inv; r[tid + 256] = e1 * inv;
    r[tid + 512] = e2 * inv; r[tid + 768] = e3 * inv;
}

// ---------------- embedding: x = tok_emb[idx] + pos_emb ----------------
__global__ void embed_kernel(const int* __restrict__ idx, const float* __restrict__ tok,
                             const float* __restrict__ pos, float* __restrict__ x) {
    long long i = (long long)blockIdx.x * blockDim.x + threadIdx.x;  // < B*T*D
    int d = (int)(i & 255);
    long long bt = i >> 8;
    int t = (int)(bt & 1023);
    int token = idx[bt];
    x[i] = tok[(long long)token * 256 + d] + pos[(long long)t * 256 + d];
}

// ---------------- mean pool over T ----------------
__global__ void pool_kernel(const float* __restrict__ xf, float* __restrict__ pooled) {
    int i = blockIdx.x * blockDim.x + threadIdx.x;  // < B*D = 2048
    int b = i >> 8, d = i & 255;
    const float* p = xf + (long long)b * 1024 * 256 + d;
    float s = 0.f;
    for (int t = 0; t < 1024; ++t) s += p[(long long)t * 256];
    pooled[i] = s * (1.f / 1024.f);
}

// ---------------- classifier ----------------
__global__ void cls1_kernel(const float* __restrict__ pooled, const float* __restrict__ W,
                            const float* __restrict__ bias, float* __restrict__ h) {
    int i = blockIdx.x * blockDim.x + threadIdx.x;  // < B*CH = 2048
    int b = i >> 8, j = i & 255;
    float s = bias[j];
    for (int d = 0; d < 256; ++d) s += pooled[b * 256 + d] * W[d * 256 + j];
    h[i] = fmaxf(s, 0.f);
}
__global__ void cls2_kernel(const float* __restrict__ h, const float* __restrict__ W,
                            const float* __restrict__ bias, float* __restrict__ out) {
    int i = threadIdx.x;
    if (i >= 16) return;
    int b = i >> 1, o = i & 1;
    float s = bias[o];
    for (int j = 0; j < 256; ++j) s += h[b * 256 + j] * W[j * 2 + o];
    out[i] = s;
}

// ---------------- host ----------------
extern "C" void kernel_launch(void* const* d_in, const int* in_sizes, int n_in,
                              void* d_out, int out_size, void* d_ws, size_t ws_size,
                              hipStream_t stream) {
    constexpr int Bsz = 8, T = 1024, D = 256, H = 4, HS = 64, FF = 1024, Lyr = 4;
    constexpr int BT = Bsz * T;  // 8192

    const int*   idx   = (const int*)  d_in[0];
    const float* tok   = (const float*)d_in[1];
    const float* pos   = (const float*)d_in[2];
    const float* Wq    = (const float*)d_in[3];
    const float* Wk    = (const float*)d_in[4];
    const float* Wv    = (const float*)d_in[5];
    const float* Wproj = (const float*)d_in[6];
    const float* bproj = (const float*)d_in[7];
    const float* ln1g  = (const float*)d_in[8];
    const float* ln1b  = (const float*)d_in[9];
    const float* ln2g  = (const float*)d_in[10];
    const float* ln2b  = (const float*)d_in[11];
    const float* W1    = (const float*)d_in[12];
    const float* b1    = (const float*)d_in[13];
    const float* W2    = (const float*)d_in[14];
    const float* b2    = (const float*)d_in[15];
    const float* lnfg  = (const float*)d_in[16];
    const float* lnfb  = (const float*)d_in[17];
    const float* cW1   = (const float*)d_in[18];
    const float* cb1   = (const float*)d_in[19];
    const float* cW2   = (const float*)d_in[20];
    const float* cb2   = (const float*)d_in[21];

    float* out  = (float*)d_out;
    float* maps = out + 16;  // [L*H, B, T, T] fp32

    // workspace carve-out
    char* w = (char*)d_ws;
    auto carve = [&](size_t bytes) -> void* {
        void* p = (void*)w;
        w += (bytes + 255) & ~(size_t)255;
        return p;
    };
    float*          x      = (float*)         carve((size_t)BT * D * 4);
    float*          xf     = (float*)         carve((size_t)BT * D * 4);
    unsigned short* xnb    = (unsigned short*)carve((size_t)BT * D * 2);
    unsigned short* Qb     = (unsigned short*)carve((size_t)BT * D * 2);
    unsigned short* Kb     = (unsigned short*)carve((size_t)BT * D * 2);
    unsigned short* Vb     = (unsigned short*)carve((size_t)BT * D * 2);
    unsigned short* Vt     = (unsigned short*)carve((size_t)BT * D * 2);
    unsigned short* Ob     = (unsigned short*)carve((size_t)BT * D * 2);
    unsigned short* ffh    = (unsigned short*)carve((size_t)BT * FF * 2);
    unsigned short* WqT    = (unsigned short*)carve((size_t)D * D * 2);
    unsigned short* WkT    = (unsigned short*)carve((size_t)D * D * 2);
    unsigned short* WvT    = (unsigned short*)carve((size_t)D * D * 2);
    unsigned short* WpT    = (unsigned short*)carve((size_t)D * D * 2);
    unsigned short* W1T    = (unsigned short*)carve((size_t)D * FF * 2);
    unsigned short* W2T    = (unsigned short*)carve((size_t)FF * D * 2);
    float*          pooled = (float*)         carve((size_t)Bsz * D * 4);
    float*          chid   = (float*)         carve((size_t)Bsz * 256 * 4);

    const long long TT  = (long long)T * T;
    const long long BTT = (long long)Bsz * TT;
    const long long TD  = (long long)T * D;

    embed_kernel<<<BT * D / 256, 256, 0, stream>>>(idx, tok, pos, x);

    for (int l = 0; l < Lyr; ++l) {
        float* mapsl = maps + (long long)l * H * BTT;

        // ---- per-layer weight prep: fp32 [K,N] -> bf16 [N,K]
        dim3 gWh(D * HS / 256, 1, H);
        wtr_kernel<<<gWh, 256, 0, stream>>>(Wq + (size_t)l * H * D * HS, WqT, D, HS,
                                            (long long)D * HS, (long long)HS * D);
        wtr_kernel<<<gWh, 256, 0, stream>>>(Wk + (size_t)l * H * D * HS, WkT, D, HS,
                                            (long long)D * HS, (long long)HS * D);
        wtr_kernel<<<gWh, 256, 0, stream>>>(Wv + (size_t)l * H * D * HS, WvT, D, HS,
                                            (long long)D * HS, (long long)HS * D);
        wtr_kernel<<<dim3(D * D / 256, 1, 1), 256, 0, stream>>>(
            Wproj + (size_t)l * D * D, WpT, D, D, 0, 0);
        wtr_kernel<<<dim3(D * FF / 256, 1, 1), 256, 0, stream>>>(
            W1 + (size_t)l * D * FF, W1T, D, FF, 0, 0);
        wtr_kernel<<<dim3(FF * D / 256, 1, 1), 256, 0, stream>>>(
            W2 + (size_t)l * FF * D, W2T, FF, D, 0, 0);

        ln_kernel<true><<<BT, 256, 0, stream>>>(x, ln1g + l * D, ln1b + l * D, xnb);

        // ---- Q/K/V projections: M=8192, N=256, K=256  (32x64 strips)
        dim3 gQKV((BT / 32) * (D / 64) / 8, 1, 1);
        gemm_ws<0, false, false, false><<<gQKV, 256, 0, stream>>>(
            xnb, D, 0, 0, WqT, D, 0, 0,
            nullptr, 0, 0, 0, Qb, D, 0, 0, nullptr, nullptr, 0, BT, D, D, 1);
        gemm_ws<0, false, false, false><<<gQKV, 256, 0, stream>>>(
            xnb, D, 0, 0, WkT, D, 0, 0,
            nullptr, 0, 0, 0, Kb, D, 0, 0, nullptr, nullptr, 0, BT, D, D, 1);
        gemm_ws<0, false, false, false><<<gQKV, 256, 0, stream>>>(
            xnb, D, 0, 0, WvT, D, 0, 0,
            nullptr, 0, 0, 0, Vb, D, 0, 0, nullptr, nullptr, 0, BT, D, D, 1);

        vt_kernel<<<BT * D / 256, 256, 0, stream>>>(Vb, Vt);

        // ---- Scores: S[b,h] = Q K^T straight into d_out maps (z1=b, z2=h)
        dim3 gS((T / 32) * (T / 64) / 8, 1, Bsz * H);
        gemm_ws<0, false, false, false><<<gS, 256, 0, stream>>>(
            Qb, D, TD, 64, Kb, D, TD, 64,
            mapsl, T, TT, BTT, nullptr, 0, 0, 0,
            nullptr, nullptr, 0, T, T, HS, H);

        softmax_kernel<<<H * Bsz * T, 256, 0, stream>>>(mapsl, 0.125f);  // HS^-0.5

        // ---- O = P @ V : A fp32 from d_out, B = Vt bf16 [64, T]
        dim3 gO((T / 32) * (HS / 64) / 8, 1, Bsz * H);
        gemm_ws<1, false, false, false><<<gO, 256, 0, stream>>>(
            mapsl, T, TT, BTT, Vt, T, (long long)H * HS * T, (long long)HS * T,
            nullptr, 0, 0, 0, Ob, D, TD, 64,
            nullptr, nullptr, 0, T, HS, T, H);

        // ---- proj + residual: x = x + O @ Wproj + bproj
        dim3 gP((BT / 32) * (D / 64) / 8, 1, 1);
        gemm_ws<0, true, false, true><<<gP, 256, 0, stream>>>(
            Ob, D, 0, 0, WpT, D, 0, 0,
            x, D, 0, 0, nullptr, 0, 0, 0,
            bproj + l * D, x, D, BT, D, D, 1);

        ln_kernel<true><<<BT, 256, 0, stream>>>(x, ln2g + l * D, ln2b + l * D, xnb);

        // ---- FFN1: relu(xn2 @ W1 + b1) -> bf16
        dim3 gF1((BT / 32) * (FF / 64) / 8, 1, 1);
        gemm_ws<0, true, true, false><<<gF1, 256, 0, stream>>>(
            xnb, D, 0, 0, W1T, D, 0, 0,
            nullptr, 0, 0, 0, ffh, FF, 0, 0,
            b1 + l * FF, nullptr, 0, BT, FF, D, 1);

        // ---- FFN2 + residual: x = x + h @ W2 + b2
        dim3 gF2((BT / 32) * (D / 64) / 8, 1, 1);
        gemm_ws<0, true, false, true><<<gF2, 256, 0, stream>>>(
            ffh, FF, 0, 0, W2T, FF, 0, 0,
            x, D, 0, 0, nullptr, 0, 0, 0,
            b2 + l * D, x, D, BT, D, FF, 1);
    }

    ln_kernel<false><<<BT, 256, 0, stream>>>(x, lnfg, lnfb, xf);
    pool_kernel<<<Bsz, 256, 0, stream>>>(xf, pooled);
    cls1_kernel<<<Bsz, 256, 0, stream>>>(pooled, cW1, cb1, chid);
    cls2_kernel<<<1, 32, 0, stream>>>(chid, cW2, cb2, out);
}
